// SCAKDloss_39788577030273
// MI455X (gfx1250) — compile-verified, matrix operand order checked by reading, not run on hardware
//
#include <hip/hip_runtime.h>
#include <hip/hip_bf16.h>
#include <math.h>

// ---------------------------------------------------------------------------
// SCAKD loss for MI455X (gfx1250, wave32, WMMA).
// Dominant cost: two 4096x4096x2048 Gram GEMMs -> v_wmma_f32_16x16x32_bf16
// with split-bf16 (hi+lo) accumulation for ~fp32 accuracy at bf16-WMMA rate.
// GEMM v2: 128x64 block tile, 32x32 wave tile (24 WMMA/iter/wave),
// fragment-major LDS layout (2x ds_load_b128 per fragment), register
// double-buffering of global loads to hide HBM latency behind WMMA.
// ---------------------------------------------------------------------------

#define NROW 4096
#define DIM  2048
#define TAUF 0.04f
#define BIGV 100000.0f

#define BM 128
#define BN 64
#define BK 32
#define FSTR 24   // bf16 elems per (row,half) fragment slot: 16 data + 8 pad (48B)

typedef __attribute__((ext_vector_type(16))) __bf16 v16bf;
typedef __attribute__((ext_vector_type(8)))  __bf16 v8bf;
typedef __attribute__((ext_vector_type(4)))  __bf16 v4bf;
typedef __attribute__((ext_vector_type(8)))  float  v8f;

// workspace layout (in floats):
//   [0..3]                      : accum (d, e, m branch sums)
//   [64 .. 64+4096)             : inv_t
//   [64+4096 .. 64+8192)        : inv_s
//   [64+8192 .. 64+12288)       : t2 (sum of normalized t^2 per row)
//   [64+12288 .. 64+16384)      : s2
//   [16448 .. 16448+16M)        : dist_t (4096x4096)
//   [16448+16M .. 16448+32M)    : dist   (4096x4096)
// total ~128.06 MB
#define WS_INV_T  64
#define WS_INV_S  (64 + NROW)
#define WS_T2     (64 + 2*NROW)
#define WS_S2     (64 + 3*NROW)
#define WS_DIST_T 16448

// ---------------------------------------------------------------------------
__global__ void __launch_bounds__(32)
zero_kernel(float* __restrict__ ws) {
    if (threadIdx.x < 4) ws[threadIdx.x] = 0.0f;
}

// ---------------------------------------------------------------------------
// Per-row norms. grid = (NROW, 2): y==0 -> teacher, y==1 -> student.
__global__ void __launch_bounds__(256)
rownorm_kernel(const float* __restrict__ T, const float* __restrict__ S,
               float* __restrict__ ws) {
    const int row   = blockIdx.x;
    const int which = blockIdx.y;
    const float* src = which ? S : T;
    float* inv_out = ws + (which ? WS_INV_S : WS_INV_T);
    float* sq_out  = ws + (which ? WS_S2    : WS_T2);

    const float* p = src + (size_t)row * DIM;
    float acc = 0.0f;
    for (int c = threadIdx.x; c < DIM; c += 256) {
        float x = p[c];
        acc += x * x;
    }
    __shared__ float red[256];
    red[threadIdx.x] = acc;
    __syncthreads();
    for (int off = 128; off > 0; off >>= 1) {
        if (threadIdx.x < off) red[threadIdx.x] += red[threadIdx.x + off];
        __syncthreads();
    }
    if (threadIdx.x == 0) {
        float n2  = red[0];
        float nrm = sqrtf(n2);
        float inv = 1.0f / fmaxf(nrm, 1e-12f);
        inv_out[row] = inv;
        sq_out[row]  = n2 * inv * inv;   // ||normalized row||^2 (== ~1)
    }
}

// ---------------------------------------------------------------------------
// Stage 4 consecutive K-values (one float4) into fragment-major LDS layout.
// Fragment element order per (row, half h): e in [0..7] -> K = e + 8h,
//                                           e in [8..15] -> K = e + 8 + 8h.
// Inverse: h = (K>>3)&1 ; e = (K&7) + (K>=16 ? 8 : 0).
// A float4 at K-base lc (multiple of 4) lands in ONE contiguous 4-elem run.
__device__ __forceinline__ void stage4(__bf16* __restrict__ hiL,
                                       __bf16* __restrict__ loL,
                                       int r, int lc, float4 v, float sc) {
    const int h   = (lc >> 3) & 1;
    const int e0  = (lc & 7) + ((lc & 16) ? 8 : 0);
    const int off = (r * 2 + h) * FSTR + e0;
    float x0 = v.x * sc, x1 = v.y * sc, x2 = v.z * sc, x3 = v.w * sc;
    __bf16 h0 = (__bf16)x0, h1 = (__bf16)x1, h2 = (__bf16)x2, h3 = (__bf16)x3;
    v4bf hv = {h0, h1, h2, h3};
    v4bf lv = {(__bf16)(x0 - (float)h0), (__bf16)(x1 - (float)h1),
               (__bf16)(x2 - (float)h2), (__bf16)(x3 - (float)h3)};
    *(v4bf*)(hiL + off) = hv;
    *(v4bf*)(loL + off) = lv;
}

// One 16x32 operand fragment = 32 contiguous bytes per lane -> 2x ds_load_b128.
__device__ __forceinline__ v16bf load_frag(const __bf16* __restrict__ layer,
                                           int r0, int lane) {
    const __bf16* p = layer + ((r0 + (lane & 15)) * 2 + (lane >> 4)) * FSTR;
    v8bf a = *(const v8bf*)p;
    v8bf b = *(const v8bf*)(p + 8);
    return __builtin_shufflevector(a, b, 0, 1, 2, 3, 4, 5, 6, 7,
                                   8, 9, 10, 11, 12, 13, 14, 15);
}

#define WMMA_BF16(C, A, B) \
    (C) = __builtin_amdgcn_wmma_f32_16x16x32_bf16(false, (A), false, (B), (short)0, (C), false, false)

// ---------------------------------------------------------------------------
// Fused Gram GEMM + distance transform.
// Rows [by*128, +128) x cols [bx*64, +64) of BOTH
//   dist_t = sqrt(max(t2_i + t2_j - 2 t_i.t_j, 1e-12))
//   dist   = sqrt(max(t2_i + s2_j - 2 t_i.s_j, 1e-12))
// sharing the A (teacher) tile between both products.
__global__ void __launch_bounds__(256)
gram_dist_kernel(const float* __restrict__ T, const float* __restrict__ S,
                 const float* __restrict__ ws,
                 float* __restrict__ dist_t, float* __restrict__ dist) {
    const float* inv_t = ws + WS_INV_T;
    const float* inv_s = ws + WS_INV_S;
    const float* t2    = ws + WS_T2;
    const float* s2    = ws + WS_S2;

    __shared__ __bf16 sAhi[BM * 2 * FSTR], sAlo[BM * 2 * FSTR];     // 12KB each
    __shared__ __bf16 sBtHi[BN * 2 * FSTR], sBtLo[BN * 2 * FSTR];   // 6KB each
    __shared__ __bf16 sBsHi[BN * 2 * FSTR], sBsLo[BN * 2 * FSTR];

    const int tid  = threadIdx.x;
    const int lane = tid & 31;
    const int wave = tid >> 5;
    const int wr   = wave >> 1;   // 0..3 -> 32-row strip
    const int wc   = wave & 1;    // 0..1 -> 32-col strip
    const int rowBase = blockIdx.y * BM;
    const int colBase = blockIdx.x * BN;

    v8f ctt[2][2] = {{{}, {}}, {{}, {}}};
    v8f cts[2][2] = {{{}, {}}, {{}, {}}};

    // loader mapping: lr in [0,32), lc in {0,4,...,28}
    const int lr = tid >> 3;
    const int lc = (tid & 7) * 4;

    float scA[4], scBt[2], scBs[2];
#pragma unroll
    for (int q = 0; q < 4; ++q) scA[q] = inv_t[rowBase + lr + 32 * q];
#pragma unroll
    for (int q = 0; q < 2; ++q) {
        scBt[q] = inv_t[colBase + lr + 32 * q];
        scBs[q] = inv_s[colBase + lr + 32 * q];
    }

    // register double-buffer of staged global data
    float4 rA[4], rB[2], rS[2];
#pragma unroll
    for (int q = 0; q < 4; ++q)
        rA[q] = *(const float4*)(T + (size_t)(rowBase + lr + 32 * q) * DIM + lc);
#pragma unroll
    for (int q = 0; q < 2; ++q) {
        rB[q] = *(const float4*)(T + (size_t)(colBase + lr + 32 * q) * DIM + lc);
        rS[q] = *(const float4*)(S + (size_t)(colBase + lr + 32 * q) * DIM + lc);
    }

    for (int kk = 0; kk < DIM; kk += BK) {
        __syncthreads();   // LDS free from previous iteration's compute
#pragma unroll
        for (int q = 0; q < 4; ++q)
            stage4(sAhi, sAlo, lr + 32 * q, lc, rA[q], scA[q]);
#pragma unroll
        for (int q = 0; q < 2; ++q) {
            stage4(sBtHi, sBtLo, lr + 32 * q, lc, rB[q], scBt[q]);
            stage4(sBsHi, sBsLo, lr + 32 * q, lc, rS[q], scBs[q]);
        }
        __syncthreads();

        // issue next iteration's global loads now; they overlap the WMMAs
        if (kk + BK < DIM) {
            const int kn = kk + BK + lc;
#pragma unroll
            for (int q = 0; q < 4; ++q)
                rA[q] = *(const float4*)(T + (size_t)(rowBase + lr + 32 * q) * DIM + kn);
#pragma unroll
            for (int q = 0; q < 2; ++q) {
                rB[q] = *(const float4*)(T + (size_t)(colBase + lr + 32 * q) * DIM + kn);
                rS[q] = *(const float4*)(S + (size_t)(colBase + lr + 32 * q) * DIM + kn);
            }
        }

        v16bf aH[2], aL[2], btH[2], btL[2], bsH[2], bsL[2];
#pragma unroll
        for (int m = 0; m < 2; ++m) {
            aH[m] = load_frag(sAhi, wr * 32 + m * 16, lane);
            aL[m] = load_frag(sAlo, wr * 32 + m * 16, lane);
        }
#pragma unroll
        for (int n = 0; n < 2; ++n) {
            btH[n] = load_frag(sBtHi, wc * 32 + n * 16, lane);
            btL[n] = load_frag(sBtLo, wc * 32 + n * 16, lane);
            bsH[n] = load_frag(sBsHi, wc * 32 + n * 16, lane);
            bsL[n] = load_frag(sBsLo, wc * 32 + n * 16, lane);
        }

        // split-bf16: hi*hi + hi*lo + lo*hi (lo*lo dropped)
#pragma unroll
        for (int m = 0; m < 2; ++m) {
#pragma unroll
            for (int n = 0; n < 2; ++n) {
                WMMA_BF16(ctt[m][n], aH[m], btH[n]);
                WMMA_BF16(ctt[m][n], aH[m], btL[n]);
                WMMA_BF16(ctt[m][n], aL[m], btH[n]);
                WMMA_BF16(cts[m][n], aH[m], bsH[n]);
                WMMA_BF16(cts[m][n], aH[m], bsL[n]);
                WMMA_BF16(cts[m][n], aL[m], bsH[n]);
            }
        }
    }

    // epilogue: C/D layout (VGPR p, half h): M = p + 8h, N = lane&15
    const int h  = lane >> 4;
    const int nl = lane & 15;
#pragma unroll
    for (int m = 0; m < 2; ++m) {
#pragma unroll
        for (int n = 0; n < 2; ++n) {
#pragma unroll
            for (int p = 0; p < 8; ++p) {
                const int i = rowBase + wr * 32 + m * 16 + p + 8 * h;
                const int j = colBase + wc * 32 + n * 16 + nl;
                const float ti2 = t2[i];
                float v0 = ti2 + t2[j] - 2.0f * ctt[m][n][p];
                dist_t[(size_t)i * NROW + j] = sqrtf(fmaxf(v0, 1e-12f));
                float w0 = ti2 + s2[j] - 2.0f * cts[m][n][p];
                dist[(size_t)i * NROW + j] = sqrtf(fmaxf(w0, 1e-12f));
            }
        }
    }
}

// ---------------------------------------------------------------------------
__device__ __forceinline__ void red_min_pair(float& v, int& ix, float* sv, int* si, int tid) {
    sv[tid] = v; si[tid] = ix; __syncthreads();
    for (int off = 128; off > 0; off >>= 1) {
        if (tid < off) {
            float v2 = sv[tid + off]; int i2 = si[tid + off];
            if (v2 < sv[tid] || (v2 == sv[tid] && i2 < si[tid])) { sv[tid] = v2; si[tid] = i2; }
        }
        __syncthreads();
    }
    v = sv[0]; ix = si[0]; __syncthreads();
}

__device__ __forceinline__ void red_max_pair(float& v, int& ix, float* sv, int* si, int tid) {
    sv[tid] = v; si[tid] = ix; __syncthreads();
    for (int off = 128; off > 0; off >>= 1) {
        if (tid < off) {
            float v2 = sv[tid + off]; int i2 = si[tid + off];
            if (v2 > sv[tid] || (v2 == sv[tid] && i2 < si[tid])) { sv[tid] = v2; si[tid] = i2; }
        }
        __syncthreads();
    }
    v = sv[0]; ix = si[0]; __syncthreads();
}

// bitonic sort of 4096 (key,idx) pairs, ascending lexicographic (key, idx)
// == stable argsort with masked entries set to +inf.
__device__ void bitonic_sort(float* key, int* idx, int tid) {
    for (int k = 2; k <= NROW; k <<= 1) {
        for (int j = k >> 1; j > 0; j >>= 1) {
            __syncthreads();
            for (int e = tid; e < NROW; e += 256) {
                int partner = e ^ j;
                if (partner > e) {
                    bool up = ((e & k) == 0);
                    float ka = key[e], kb = key[partner];
                    int ia = idx[e],  ib = idx[partner];
                    bool aGtB = (ka > kb) || (ka == kb && ia > ib);
                    if (aGtB == up) {
                        key[e] = kb; key[partner] = ka;
                        idx[e] = ib; idx[partner] = ia;
                    }
                }
            }
        }
    }
    __syncthreads();
}

__device__ __forceinline__ float pair_ce(float an_v, float an_t, float ap_v, float ap_t) {
    float w_an = fmaxf(an_t - an_v, 0.0f);
    float w_ap = fmaxf(ap_v - ap_t, 0.0f);
    float l0 = w_an * an_v / TAUF;
    float l1 = w_ap * ap_v / TAUF;
    float mm = fmaxf(l0, l1);
    float lse = mm + logf(expf(l0 - mm) + expf(l1 - mm));
    return lse - l0;
}

// one block per row: masked extremes + masked lower-medians + CE accumulation
__global__ void __launch_bounds__(256)
mining_kernel(const float* __restrict__ dist, const float* __restrict__ dist_t,
              float* __restrict__ accum) {
    const int i   = blockIdx.x;
    const int tid = threadIdx.x;

    __shared__ float    s_dist[NROW];        // 16 KB: cached dist row
    __shared__ float    s_key[NROW];         // 16 KB: sort keys
    __shared__ int      s_sidx[NROW];        // 16 KB: sort payload (orig idx)
    __shared__ unsigned s_mask[NROW / 32];   // neg-mask bits
    __shared__ float    s_rv[256];
    __shared__ int      s_ri[256];

    const float* drow = dist   + (size_t)i * NROW;
    const float* trow = dist_t + (size_t)i * NROW;
    const float temp1 = drow[i];

    // pass 1: cache row, build mask via wave32 ballot, thread-local extremes
    float anV  = BIGV; int anI  = NROW;   // argmin over where(neg, d, BIG)
    float apV  = 0.f;  int apI  = NROW;   // argmax over where(pos, d, 0)
    float aneV = 0.f;  int aneI = NROW;   // argmax over where(neg, d, 0)
    float apeV = BIGV; int apeI = NROW;   // argmin over where(pos, d, BIG)
    int cnt = 0;
    for (int j = tid; j < NROW; j += 256) {
        float dv = drow[j];
        float tv = trow[j];
        bool neg = tv > temp1;
        s_dist[j] = dv;
        unsigned long long bal = __ballot(neg);
        if ((tid & 31) == 0) s_mask[j >> 5] = (unsigned)bal;
        cnt += neg ? 1 : 0;
        float c;
        c = neg ? dv : BIGV; if (c < anV  || (c == anV  && j < anI))  { anV  = c; anI  = j; }
        c = neg ? dv : 0.f;  if (c > aneV || (c == aneV && j < aneI)) { aneV = c; aneI = j; }
        c = neg ? BIGV : dv; if (c < apeV || (c == apeV && j < apeI)) { apeV = c; apeI = j; }
        c = neg ? 0.f : dv;  if (c > apV  || (c == apV  && j < apI))  { apV  = c; apI  = j; }
    }

    // neg count
    s_ri[tid] = cnt; __syncthreads();
    for (int off = 128; off > 0; off >>= 1) {
        if (tid < off) s_ri[tid] += s_ri[tid + off];
        __syncthreads();
    }
    const int kneg = s_ri[0];
    const int kpos = NROW - kneg;
    __syncthreads();

    red_min_pair(anV, anI, s_rv, s_ri, tid);
    red_max_pair(apV, apI, s_rv, s_ri, tid);
    red_max_pair(aneV, aneI, s_rv, s_ri, tid);
    red_min_pair(apeV, apeI, s_rv, s_ri, tid);

    // median over neg
    for (int j = tid; j < NROW; j += 256) {
        bool neg = (s_mask[j >> 5] >> (j & 31)) & 1u;
        s_key[j]  = neg ? s_dist[j] : __builtin_inff();
        s_sidx[j] = j;
    }
    bitonic_sort(s_key, s_sidx, tid);
    int pn = (kneg - 1) / 2; if (pn < 0) pn = 0;
    const int   anmI = s_sidx[pn];
    const float anmV = (kneg > 0) ? s_dist[anmI] : 0.0f;
    __syncthreads();

    // median over pos
    for (int j = tid; j < NROW; j += 256) {
        bool neg = (s_mask[j >> 5] >> (j & 31)) & 1u;
        s_key[j]  = neg ? __builtin_inff() : s_dist[j];
        s_sidx[j] = j;
    }
    bitonic_sort(s_key, s_sidx, tid);
    int pp = (kpos - 1) / 2; if (pp < 0) pp = 0;
    const int   apmI = s_sidx[pp];
    const float apmV = (kpos > 0) ? s_dist[apmI] : 0.0f;

    if (tid == 0) {
        float ce_d = pair_ce(anV,  trow[anI],  apV,  trow[apI]);
        float ce_e = pair_ce(aneV, trow[aneI], apeV, trow[apeI]);
        float ce_m = pair_ce(anmV, trow[anmI], apmV, trow[apmI]);
        atomicAdd(&accum[0], ce_d);
        atomicAdd(&accum[1], ce_e);
        atomicAdd(&accum[2], ce_m);
    }
}

// ---------------------------------------------------------------------------
__global__ void __launch_bounds__(32)
finalize_kernel(const float* __restrict__ accum, const int* __restrict__ epoch,
                float* __restrict__ out) {
    if (threadIdx.x == 0 && blockIdx.x == 0) {
        const float pi = 3.14159265358979323846f;
        float step = (float)epoch[0];
        float de = 0.5f * 0.75f * (1.0f + cosf(pi * step / 120.0f));
        float dm = 0.5f * 0.75f * (1.0f + cosf(pi * step / 240.0f));
        float e = (step < 120.0f) ? de : 0.0f;
        float m = dm;
        float d = 3.0f - e - m;
        float inv_n = 1.0f / (float)NROW;
        out[0] = accum[1] * inv_n * e + accum[2] * inv_n * m + accum[0] * inv_n * d;
    }
}

// ---------------------------------------------------------------------------
extern "C" void kernel_launch(void* const* d_in, const int* in_sizes, int n_in,
                              void* d_out, int out_size, void* d_ws, size_t ws_size,
                              hipStream_t stream) {
    const float* T     = (const float*)d_in[0];   // teacher_inputs [4096,2048]
    const float* S     = (const float*)d_in[1];   // inputs         [4096,2048]
    const int*   epoch = (const int*)d_in[2];     // scalar

    float* ws     = (float*)d_ws;
    float* dist_t = ws + WS_DIST_T;
    float* dist   = dist_t + (size_t)NROW * NROW;
    float* out    = (float*)d_out;

    zero_kernel<<<1, 32, 0, stream>>>(ws);
    rownorm_kernel<<<dim3(NROW, 2), 256, 0, stream>>>(T, S, ws);
    gram_dist_kernel<<<dim3(NROW / BN, NROW / BM), 256, 0, stream>>>(T, S, ws, dist_t, dist);
    mining_kernel<<<NROW, 256, 0, stream>>>(dist, dist_t, ws);
    finalize_kernel<<<1, 32, 0, stream>>>(ws, epoch, out);
}
